// RNN_encoder_decoder_7181185319029
// MI455X (gfx1250) — matrix operand order repaired and checked by reading, not compiled, not run on hardware
//
#include <hip/hip_runtime.h>
#include <hip/hip_bf16.h>

// ---------------------------------------------------------------------------
// Seq2seq tanh-RNN (B=256, S=256, I=3, H=1024, T=64) on gfx1250.
// Core op per time step:  Hnew = tanh( [inp | Hprev] @ [Wih;Whh]^T + b )
// fp16 WMMA GEMM (v_wmma_f32_16x16x32_f16, fp32 accum) fed by a depth-2
// async-to-LDS pipeline (global_load_async_to_lds_b128 / s_wait_asynccnt).
// ---------------------------------------------------------------------------

#define BATCH 256
#define SEQ   256
#define IDIM  3
#define HDIM  1024
#define TLEN  64

typedef __attribute__((ext_vector_type(16))) _Float16 v16h;
typedef __attribute__((ext_vector_type(8)))  _Float16 v8h;
typedef __attribute__((ext_vector_type(8)))  float    v8f;

// LDS tile pitch: 32 k-halves + 8 pad halves = 80B rows (bank-conflict free,
// 16B aligned for ds_load_b128).
#define LPITCH 40

// ------------------------- utility kernels ---------------------------------

__global__ void f32_to_f16_kernel(const float* __restrict__ s,
                                  _Float16* __restrict__ d, int n) {
    int i = blockIdx.x * blockDim.x + threadIdx.x;
    if (i < n) d[i] = (_Float16)s[i];
}

__global__ void combine_bias_kernel(const float* __restrict__ a,
                                    const float* __restrict__ b,
                                    float* __restrict__ o, int n) {
    int i = blockIdx.x * blockDim.x + threadIdx.x;
    if (i < n) o[i] = a[i] + b[i];
}

__global__ void fill_zero_f16_kernel(_Float16* __restrict__ p, int n) {
    int i = blockIdx.x * blockDim.x + threadIdx.x;
    if (i < n) p[i] = (_Float16)0.0f;
}

// dec_input0 = x[:, S-1, :]
__global__ void init_dec_inp_kernel(const float* __restrict__ x,
                                    float* __restrict__ inp) {
    int i = blockIdx.x * blockDim.x + threadIdx.x;
    if (i < BATCH * IDIM) {
        int b = i / IDIM, j = i % IDIM;
        inp[i] = x[(b * SEQ + (SEQ - 1)) * IDIM + j];
    }
}

// --------------------- async copy helpers (CDNA5) ---------------------------

__device__ __forceinline__ void async_copy16(const _Float16* g, _Float16* l) {
    // LDS byte address = low 32 bits of the generic address (ISA 10.2).
    unsigned loff = (unsigned)(unsigned long long)l;
    asm volatile("global_load_async_to_lds_b128 %0, %1, off"
                 :: "v"(loff), "v"(g) : "memory");
}

// Stage one k-tile (A: 16x64B, W: 64x64B) into per-wave LDS buffers.
// 10 async ops per call (8 for W, 2 for A); lane l moves 16B: row j*8+(l>>2),
// byte offset (l&3)*16 within the 64B row chunk.
__device__ __forceinline__ void stage_tile(const _Float16* __restrict__ As,
                                           const _Float16* __restrict__ Ws,
                                           int kBase,
                                           _Float16* lA, _Float16* lB,
                                           int mBase, int nBase, int lane) {
    const int r4 = lane >> 2;        // 0..7
    const int c4 = (lane & 3) * 8;   // halves
#pragma unroll
    for (int j = 0; j < 8; ++j) {
        const int row = j * 8 + r4;
        async_copy16(Ws + (nBase + row) * HDIM + kBase + c4, lB + row * LPITCH + c4);
    }
#pragma unroll
    for (int j = 0; j < 2; ++j) {
        const int row = j * 8 + r4;
        async_copy16(As + (mBase + row) * HDIM + kBase + c4, lA + row * LPITCH + c4);
    }
}

// ------------------------- fused RNN step GEMM ------------------------------
// Hout(BATCH x HDIM) = tanh( A @ W^T + bias [+ x3 @ W3^T] ) where
//   A = [A0 | A1] (BATCH x KT*32 fp16), W = [W0 ; W1] (HDIM x KT*32 fp16)
//   KT = 32 (K=1024) or 64 (K=2048, source switch at the K=1024 boundary)
// Grid: 64 blocks x 128 threads (4 waves). Each wave owns a 16x64 output
// strip and pipelines its tiles through LDS with ASYNCcnt depth 2.

template <int KT>
__global__ __launch_bounds__(128)
void rnn_gemm_tanh_kernel(const _Float16* __restrict__ A0,
                          const _Float16* __restrict__ A1,
                          const _Float16* __restrict__ W0,
                          const _Float16* __restrict__ W1,
                          const float*    __restrict__ bias,
                          const float*    __restrict__ x3, int x3_stride,
                          const float*    __restrict__ W3,
                          _Float16*       __restrict__ Hout) {
    __shared__ __align__(16) _Float16 ldsB[4][2][64 * LPITCH]; // 40 KB
    __shared__ __align__(16) _Float16 ldsA[4][2][16 * LPITCH]; // 10 KB

    const int lane  = threadIdx.x & 31;
    const int wave  = threadIdx.x >> 5;        // 0..3
    const int gw    = blockIdx.x * 4 + wave;   // 0..255 global wave id
    const int mBase = (gw >> 4) * 16;          // 16 M-tiles cover BATCH=256
    const int nBase = (gw & 15) * 64;          // 16 quads  cover HDIM=1024

    const int half  = lane >> 4;               // lane group (0/1)
    const int mRow  = lane & 15;
    const int nCol  = lane & 15;

    // ---- prologue: stage k-tiles 0 and 1 ----
    {
        const _Float16* As = A0;  // kt=0 always uses the first source
        stage_tile(As, W0, 0, &ldsA[wave][0][0], &ldsB[wave][0][0], mBase, nBase, lane);
        if (KT > 1) {
            const _Float16* As1 = (1 < 32) ? A0 : A1;
            const _Float16* Ws1 = (1 < 32) ? W0 : W1;
            stage_tile(As1, Ws1, (1 & 31) * 32,
                       &ldsA[wave][1][0], &ldsB[wave][1][0], mBase, nBase, lane);
        }
    }

    v8f acc[4] = {};

#pragma unroll
    for (int kt = 0; kt < KT; ++kt) {
        const int cur = kt & 1;

        // tile kt resident when <=10 async ops remain (only tile kt+1's ops);
        // async ops retire in order.
        if (kt + 1 < KT) asm volatile("s_wait_asynccnt 0xa" ::: "memory");
        else             asm volatile("s_wait_asynccnt 0x0" ::: "memory");

        // ---- A fragment from LDS (16x32 f16, ISA 7.12.2 layout) ----
        const _Float16* la = &ldsA[wave][cur][mRow * LPITCH + half * 8];
        v8h aLo = *(const v8h*)(la);
        v8h aHi = *(const v8h*)(la + 16);
        v16h a;
#pragma unroll
        for (int i = 0; i < 8; ++i) { a[i] = aLo[i]; a[8 + i] = aHi[i]; }

        // ---- B fragments from LDS (32x16 f16 = W^T rows) ----
        v16h bfrag[4];
#pragma unroll
        for (int nt = 0; nt < 4; ++nt) {
            const _Float16* lb =
                &ldsB[wave][cur][(nt * 16 + nCol) * LPITCH + half * 8];
            v8h blo = *(const v8h*)(lb);
            v8h bhi = *(const v8h*)(lb + 16);
#pragma unroll
            for (int i = 0; i < 8; ++i) { bfrag[nt][i] = blo[i]; bfrag[nt][8 + i] = bhi[i]; }
        }

#pragma unroll
        for (int nt = 0; nt < 4; ++nt)
            acc[nt] = __builtin_amdgcn_wmma_f32_16x16x32_f16(
                false, a, false, bfrag[nt], (short)0, acc[nt], false, false);

        // ---- re-stage this buffer with tile kt+2 (ds reads are complete:
        // the WMMAs above already forced the dscnt wait) ----
        if (kt + 2 < KT) {
            asm volatile("s_wait_dscnt 0x0" ::: "memory");
            const int kn = kt + 2;
            const _Float16* As = (kn < 32) ? A0 : A1;
            const _Float16* Ws = (kn < 32) ? W0 : W1;
            stage_tile(As, Ws, (kn & 31) * 32,
                       &ldsA[wave][cur][0], &ldsB[wave][cur][0], mBase, nBase, lane);
        }

        // ---- deep prefetch (HBM -> L2), distance 6 k-tiles ----
        if (kt + 6 < KT) {
            const int kp = kt + 6;
            const _Float16* As = (kp < 32) ? A0 : A1;
            const _Float16* Ws = (kp < 32) ? W0 : W1;
            const int kB = (kp & 31) * 32;
            __builtin_prefetch(Ws + (nBase + lane) * HDIM + kB, 0, 1);
            __builtin_prefetch(Ws + (nBase + 32 + lane) * HDIM + kB, 0, 1);
            __builtin_prefetch(As + (mBase + mRow) * HDIM + kB, 0, 1);
        }
    }

    // ---- epilogue: bias (+ K=3 input projection) + tanh, write fp16 state.
    // C/D layout: element (VGPR r, lane) -> M = mBase + half*8 + r, N = nCol.
    const int mOut = mBase + half * 8;
#pragma unroll
    for (int nt = 0; nt < 4; ++nt) {
        const int n = nBase + nt * 16 + nCol;
        const float bn = bias[n];
        float w30 = 0.f, w31 = 0.f, w32 = 0.f;
        if (W3) { w30 = W3[n * 3 + 0]; w31 = W3[n * 3 + 1]; w32 = W3[n * 3 + 2]; }
#pragma unroll
        for (int r = 0; r < 8; ++r) {
            const int m = mOut + r;
            float v = acc[nt][r] + bn;
            if (x3) {
                const float* xr = x3 + m * x3_stride;
                v += xr[0] * w30 + xr[1] * w31 + xr[2] * w32;
            }
            v = tanhf(v);
            Hout[m * HDIM + n] = (_Float16)v;
        }
    }
}

// ------------------------- decoder output step ------------------------------
// out[b] = hB[b,:] . Wlin + blin ; writes d_out[b,t]; computes next dec input:
// c0 = out, c1 = inp0 - c0, c2 = inp1 - c1.  One wave per batch row.

__global__ __launch_bounds__(256)
void dec_out_kernel(const _Float16* __restrict__ hB,
                    const float* __restrict__ Wlin,
                    const float* __restrict__ blin,
                    const float* __restrict__ inp_old,
                    float* __restrict__ inp_new,
                    float* __restrict__ out, int t) {
    const int lane = threadIdx.x & 31;
    const int wave = threadIdx.x >> 5;
    const int b    = blockIdx.x * 8 + wave;  // 32 blocks x 8 waves = 256 rows

    float s = 0.f;
    for (int i = lane; i < HDIM; i += 32)
        s += (float)hB[b * HDIM + i] * Wlin[i];
#pragma unroll
    for (int off = 16; off > 0; off >>= 1)
        s += __shfl_down(s, off, 32);

    if (lane == 0) {
        const float c0 = s + blin[0];
        out[b * TLEN + t] = c0;
        const float c1 = inp_old[b * 3 + 0] - c0;
        const float c2 = inp_old[b * 3 + 1] - c1;
        inp_new[b * 3 + 0] = c0;
        inp_new[b * 3 + 1] = c1;
        inp_new[b * 3 + 2] = c2;
    }
}

// ------------------------------ host side -----------------------------------

extern "C" void kernel_launch(void* const* d_in, const int* in_sizes, int n_in,
                              void* d_out, int out_size, void* d_ws, size_t ws_size,
                              hipStream_t stream) {
    const float* x        = (const float*)d_in[0];
    const float* enc_Wih0 = (const float*)d_in[2];
    const float* enc_Whh0 = (const float*)d_in[3];
    const float* enc_bih0 = (const float*)d_in[4];
    const float* enc_bhh0 = (const float*)d_in[5];
    const float* enc_Wih1 = (const float*)d_in[6];
    const float* enc_Whh1 = (const float*)d_in[7];
    const float* enc_bih1 = (const float*)d_in[8];
    const float* enc_bhh1 = (const float*)d_in[9];
    const float* dec_Wih0 = (const float*)d_in[10];
    const float* dec_Whh0 = (const float*)d_in[11];
    const float* dec_bih0 = (const float*)d_in[12];
    const float* dec_bhh0 = (const float*)d_in[13];
    const float* dec_Wih1 = (const float*)d_in[14];
    const float* dec_Whh1 = (const float*)d_in[15];
    const float* dec_bih1 = (const float*)d_in[16];
    const float* dec_bhh1 = (const float*)d_in[17];
    const float* dec_Wlin = (const float*)d_in[18];
    const float* dec_blin = (const float*)d_in[19];

    // ---- carve workspace (256B-aligned chunks) ----
    char* ws = (char*)d_ws;
    auto carve = [&](size_t bytes) -> char* {
        char* p = ws;
        ws += (bytes + 255) & ~(size_t)255;
        return p;
    };
    const size_t WBYTES = (size_t)HDIM * HDIM * sizeof(_Float16);
    const size_t SBYTES = (size_t)BATCH * HDIM * sizeof(_Float16);

    _Float16* Whh0h  = (_Float16*)carve(WBYTES);
    _Float16* Wih1h  = (_Float16*)carve(WBYTES);
    _Float16* Whh1h  = (_Float16*)carve(WBYTES);
    _Float16* dWhh0h = (_Float16*)carve(WBYTES);
    _Float16* dWih1h = (_Float16*)carve(WBYTES);
    _Float16* dWhh1h = (_Float16*)carve(WBYTES);

    _Float16* zero16   = (_Float16*)carve(SBYTES);
    _Float16* y0b[2]   = {(_Float16*)carve(SBYTES), (_Float16*)carve(SBYTES)};
    _Float16* h1b[2]   = {(_Float16*)carve(SBYTES), (_Float16*)carve(SBYTES)};
    _Float16* hAb[2]   = {(_Float16*)carve(SBYTES), (_Float16*)carve(SBYTES)};
    _Float16* hBb[2]   = {(_Float16*)carve(SBYTES), (_Float16*)carve(SBYTES)};

    float* benc0 = (float*)carve(HDIM * sizeof(float));
    float* benc1 = (float*)carve(HDIM * sizeof(float));
    float* bdec0 = (float*)carve(HDIM * sizeof(float));
    float* bdec1 = (float*)carve(HDIM * sizeof(float));
    float* inpb[2] = {(float*)carve(BATCH * IDIM * sizeof(float)),
                      (float*)carve(BATCH * IDIM * sizeof(float))};

    // ---- one-time prep (deterministic, re-done every call) ----
    const int NW = HDIM * HDIM;
    f32_to_f16_kernel<<<(NW + 255) / 256, 256, 0, stream>>>(enc_Whh0, Whh0h, NW);
    f32_to_f16_kernel<<<(NW + 255) / 256, 256, 0, stream>>>(enc_Wih1, Wih1h, NW);
    f32_to_f16_kernel<<<(NW + 255) / 256, 256, 0, stream>>>(enc_Whh1, Whh1h, NW);
    f32_to_f16_kernel<<<(NW + 255) / 256, 256, 0, stream>>>(dec_Whh0, dWhh0h, NW);
    f32_to_f16_kernel<<<(NW + 255) / 256, 256, 0, stream>>>(dec_Wih1, dWih1h, NW);
    f32_to_f16_kernel<<<(NW + 255) / 256, 256, 0, stream>>>(dec_Whh1, dWhh1h, NW);

    combine_bias_kernel<<<(HDIM + 255) / 256, 256, 0, stream>>>(enc_bih0, enc_bhh0, benc0, HDIM);
    combine_bias_kernel<<<(HDIM + 255) / 256, 256, 0, stream>>>(enc_bih1, enc_bhh1, benc1, HDIM);
    combine_bias_kernel<<<(HDIM + 255) / 256, 256, 0, stream>>>(dec_bih0, dec_bhh0, bdec0, HDIM);
    combine_bias_kernel<<<(HDIM + 255) / 256, 256, 0, stream>>>(dec_bih1, dec_bhh1, bdec1, HDIM);

    const int NS = BATCH * HDIM;
    fill_zero_f16_kernel<<<(NS + 255) / 256, 256, 0, stream>>>(zero16, NS);
    init_dec_inp_kernel<<<(BATCH * IDIM + 255) / 256, 256, 0, stream>>>(x, inpb[0]);

    // ---- encoder: layer0 and layer1 interleaved per time step ----
    const _Float16* h0prev = zero16;
    const _Float16* h1prev = zero16;
    for (int t = 0; t < SEQ; ++t) {
        _Float16* h0next = y0b[t & 1];
        // layer0: K=1024 (Whh0), + K=3 input projection from x[:, t, :]
        rnn_gemm_tanh_kernel<32><<<64, 128, 0, stream>>>(
            h0prev, nullptr, Whh0h, nullptr, benc0,
            x + (size_t)t * IDIM, SEQ * IDIM, enc_Wih0, h0next);
        _Float16* h1next = h1b[t & 1];
        // layer1: K=2048 -> [y0_t | h1prev] @ [Wih1;Whh1]^T
        rnn_gemm_tanh_kernel<64><<<64, 128, 0, stream>>>(
            h0next, h1prev, Wih1h, Whh1h, benc1,
            nullptr, 0, nullptr, h1next);
        h0prev = h0next;
        h1prev = h1next;
    }

    // ---- decoder: hA <- enc layer0 final, hB <- enc layer1 final ----
    const _Float16* hA = h0prev;
    const _Float16* hB = h1prev;
    const float* inp_cur = inpb[0];
    for (int t = 0; t < TLEN; ++t) {
        _Float16* hAn = hAb[t & 1];
        rnn_gemm_tanh_kernel<32><<<64, 128, 0, stream>>>(
            hA, nullptr, dWhh0h, nullptr, bdec0,
            inp_cur, IDIM, dec_Wih0, hAn);
        _Float16* hBn = hBb[t & 1];
        rnn_gemm_tanh_kernel<64><<<64, 128, 0, stream>>>(
            hAn, hB, dWih1h, dWhh1h, bdec1,
            nullptr, 0, nullptr, hBn);
        float* inp_nxt = inpb[(t + 1) & 1];
        dec_out_kernel<<<32, 256, 0, stream>>>(
            hBn, dec_Wlin, dec_blin, inp_cur, inp_nxt, (float*)d_out, t);
        hA = hAn;
        hB = hBn;
        inp_cur = inp_nxt;
    }
}
